// GraphAttention_77154792505514
// MI455X (gfx1250) — compile-verified, matrix operand order checked by reading, not run on hardware
//
#include <hip/hip_runtime.h>

// Problem constants (fixed by the reference file).
#define B_   8
#define N_   4096
#define EPN  17            // DEG + 1 edges per source node
#define F_   64            // F_OUT
#define NEF  4096          // NE_FEAT (node values start at x[:, NEF])
#define E_   (N_ * EPN)    // 69632 edges
#define XROW (NEF + N_)    // 8192 floats per batch row of x

typedef __attribute__((ext_vector_type(2)))  float    v2f;
typedef __attribute__((ext_vector_type(8)))  float    v8f;
typedef __attribute__((ext_vector_type(16))) _Float16 v16h;

__device__ __forceinline__ float lrelu(float v) { return v > 0.0f ? v : 0.01f * v; }

// ---------------------------------------------------------------------------
// Kernel 1: per-batch edge scores + softmax normalization.
// One 256-thread block per batch. Writes exp(lrelu(e) - max) into ws_e and the
// softmax denominator into ws_denom[b].
// ---------------------------------------------------------------------------
__global__ __launch_bounds__(256) void gat_edge_softmax(
    const float* __restrict__ x, const float* __restrict__ W,
    const float* __restrict__ a, const int* __restrict__ src,
    const int* __restrict__ dst, float* __restrict__ ws_e,
    float* __restrict__ ws_denom)
{
    __shared__ float s_s1, s_s2, s_m;
    __shared__ float s_red[256];

    const int b   = blockIdx.x;
    const int tid = threadIdx.x;

    if (tid == 0) {
        // s1 = W . a[:64], s2 = W . a[64:]  (rank-1 collapse of the einsums)
        float s1 = 0.0f, s2 = 0.0f;
        for (int f = 0; f < F_; ++f) {
            float w = W[f];
            s1 += w * a[f];
            s2 += w * a[F_ + f];
        }
        s_s1 = s1; s_s2 = s2;
    }
    __syncthreads();

    const float  s1   = s_s1, s2 = s_s2;
    const float* node = x + (size_t)b * XROW + NEF;
    float*       eb   = ws_e + (size_t)b * E_;

    // Pass 1: e = lrelu(node[src]*s1 + node[dst]*s2), track max.
    float mloc = -1.0e30f;
    for (int j = tid; j < E_; j += 256) {
        float ev = node[src[j]] * s1 + node[dst[j]] * s2;
        ev = lrelu(ev);
        eb[j] = ev;
        mloc = fmaxf(mloc, ev);
    }
    s_red[tid] = mloc;
    __syncthreads();
    for (int off = 128; off > 0; off >>= 1) {
        if (tid < off) s_red[tid] = fmaxf(s_red[tid], s_red[tid + off]);
        __syncthreads();
    }
    if (tid == 0) s_m = s_red[0];
    __syncthreads();
    const float m = s_m;

    // Pass 2: overwrite with exp(e - m) (each thread re-reads its own writes),
    // accumulate the softmax denominator.
    float sloc = 0.0f;
    for (int j = tid; j < E_; j += 256) {
        float v = __expf(eb[j] - m);
        eb[j] = v;
        sloc += v;
    }
    __syncthreads();
    s_red[tid] = sloc;
    __syncthreads();
    for (int off = 128; off > 0; off >>= 1) {
        if (tid < off) s_red[tid] += s_red[tid + off];
        __syncthreads();
    }
    if (tid == 0) ws_denom[b] = s_red[0];
}

// ---------------------------------------------------------------------------
// Kernel 2: neighborhood aggregation + rank-1 outer product via WMMA.
// One wave32 per (batch, 16-node tile). Lane r (and its mirror r+16) computes
//   s[i] = (1/denom) * sum_k alpha[i*17+k] * node[(i+k) % N]
// then 4x v_wmma_f32_16x16x4_f32 compute the 16x64 tile  D = s ⊗ W
// (A: K=0 column holds s, B: K=0 row holds 16 W values; other K slots zero).
// ---------------------------------------------------------------------------
__global__ __launch_bounds__(32) void gat_aggregate_wmma(
    const float* __restrict__ x, const float* __restrict__ W,
    const float* __restrict__ ws_e, const float* __restrict__ ws_denom,
    float* __restrict__ out)
{
    const int blk  = blockIdx.x;
    const int b    = blk >> 8;          // / (N/16 = 256)
    const int tile = blk & 255;
    const int lane = threadIdx.x;
    const int r    = lane & 15;
    const int i    = tile * 16 + r;     // node handled by this lane (dup on hi half)

    const float* node = x + (size_t)b * XROW + NEF;
    const float* eb   = ws_e + (size_t)b * E_;

    float acc = 0.0f;
    const int base = i * EPN;
#pragma unroll
    for (int k = 0; k < EPN; ++k) {
        acc += eb[base + k] * node[(i + k) & (N_ - 1)];
    }
    acc *= 1.0f / ws_denom[b];

    const bool lo = (lane < 16);        // lanes 0-15 carry K=0 data; no branches
    float* orow = out + ((size_t)(b * N_ + tile * 16)) * F_;
    const int mbase = (lane >> 4) * 8;  // D rows: M = rr + 8*(lane>=16)
    const int ncol  = lane & 15;        // D col within 16-wide slice

#if __has_builtin(__builtin_amdgcn_wmma_f32_16x16x4_f32)
    // A: 16x4 f32, VGPR0 = K=0 (lanes 0-15) / K=2 (lanes 16-31); only K=0 set.
    v2f A;
    A.x = lo ? acc : 0.0f;
    A.y = 0.0f;
#pragma unroll
    for (int nt = 0; nt < 4; ++nt) {
        v2f Bv;
        Bv.x = lo ? W[nt * 16 + r] : 0.0f;   // B row K=0, cols nt*16..nt*16+15
        Bv.y = 0.0f;
        v8f C = {};
        v8f D = __builtin_amdgcn_wmma_f32_16x16x4_f32(
            /*neg_a=*/false, A, /*neg_b=*/false, Bv,
            /*c_mod=*/(short)0, C, /*reuse_a=*/false, /*reuse_b=*/false);
        const int col = nt * 16 + ncol;
#pragma unroll
        for (int rr = 0; rr < 8; ++rr) {
            orow[(size_t)(mbase + rr) * F_ + col] = lrelu(D[rr]);
        }
    }
#else
    // Fallback: f16 16x16x32 WMMA (K=0 slot only) -> still rank-1 outer product.
    v16h A = {};
    A[0] = (_Float16)(lo ? acc : 0.0f);
#pragma unroll
    for (int nt = 0; nt < 4; ++nt) {
        v16h Bv = {};
        Bv[0] = (_Float16)(lo ? W[nt * 16 + r] : 0.0f);
        v8f C = {};
        v8f D = __builtin_amdgcn_wmma_f32_16x16x32_f16(
            false, A, false, Bv, (short)0, C, false, false);
        const int col = nt * 16 + ncol;
#pragma unroll
        for (int rr = 0; rr < 8; ++rr) {
            orow[(size_t)(mbase + rr) * F_ + col] = lrelu(D[rr]);
        }
    }
#endif
}

// ---------------------------------------------------------------------------
// Launcher. Inputs: x(f32), W(f32), a(f32), src(i32), dst(i32), num_edges(i32).
// Workspace: [0..7] softmax denoms, [16 ..) B*E edge buffer (~2.23 MB).
// ---------------------------------------------------------------------------
extern "C" void kernel_launch(void* const* d_in, const int* in_sizes, int n_in,
                              void* d_out, int out_size, void* d_ws, size_t ws_size,
                              hipStream_t stream) {
    (void)in_sizes; (void)n_in; (void)out_size; (void)ws_size;
    const float* x   = (const float*)d_in[0];
    const float* W   = (const float*)d_in[1];
    const float* a   = (const float*)d_in[2];
    const int*   src = (const int*)d_in[3];
    const int*   dst = (const int*)d_in[4];
    float*       out = (float*)d_out;

    float* wsf      = (float*)d_ws;
    float* ws_denom = wsf;        // 8 floats
    float* ws_e     = wsf + 16;   // B*E floats

    gat_edge_softmax<<<B_, 256, 0, stream>>>(x, W, a, src, dst, ws_e, ws_denom);
    gat_aggregate_wmma<<<B_ * (N_ / 16), 32, 0, stream>>>(x, W, ws_e, ws_denom, out);
}
